// Encoder_10634339025374
// MI455X (gfx1250) — matrix operand order repaired and checked by reading, not compiled, not run on hardware
//
#include <hip/hip_runtime.h>
#include <hip/hip_fp16.h>

typedef __attribute__((ext_vector_type(16))) _Float16 v16h;
typedef __attribute__((ext_vector_type(8)))  _Float16 v8h;
typedef __attribute__((ext_vector_type(8)))  float    v8f;

#define D_MODEL 128
#define WDIM    32
#define TLEN    127
#define BATCH   4
#define DI      256
#define DSTATE  64
#define DTRANK  8
#define LSEQ    (TLEN*WDIM)      // 4064
#define MTOK    (BATCH*LSEQ)     // 16256

// ---------------- activations ----------------
__device__ __forceinline__ float act_silu(float v)     { return v / (1.f + __expf(-v)); }
__device__ __forceinline__ float act_gelu(float v)     { return 0.5f * v * (1.f + erff(v * 0.70710678118654752f)); }
__device__ __forceinline__ float act_softplus(float v) { return fmaxf(v, 0.f) + log1pf(__expf(-fabsf(v))); }

// ---------------- WMMA GEMM ----------------
// C[M,N] = act(A[M,K] @ B[K,N] + bias) (+resid)
// A  : f16 row-major, leading dim lda (mult of 8)
// Bt : f16 *transposed* weights, Bt[n][k], leading dim ldb = Kpad (mult of 8)
// C  : f32 (ldc), C16 optional f16 mirror
// ACT: 0 none, 2 gelu, 3 softplus
template<int ACT, bool HAS_BIAS, bool HAS_RESID, bool HAS_C, bool HAS_C16>
__global__ __launch_bounds__(128)
void gemm_wmma_f16(const _Float16* __restrict__ A, const _Float16* __restrict__ Bt,
                   const float* __restrict__ bias, const float* __restrict__ resid,
                   float* __restrict__ C, _Float16* __restrict__ C16,
                   int M, int N, int K, int lda, int ldb, int ldc)
{
    const int lane = threadIdx.x & 31;
    const int wave = threadIdx.x >> 5;
    const int m0 = (blockIdx.y * 4 + wave) * 16;
    const int n0 = blockIdx.x * 16;
    if (m0 >= M) return;
    const int lo = lane & 15, hi = lane >> 4;

    v8f acc = {};
    const _Float16* arow = A  + (long)(m0 + lo) * lda;
    const _Float16* brow = Bt + (long)(n0 + lo) * ldb;
    for (int kk = 0; kk < K; kk += 32) {
        __builtin_prefetch(arow + kk + 128, 0, 1);
        __builtin_prefetch(brow + kk + 128, 0, 1);
        // A frag: lanes 0-15 rows m0..m0+15 / K {kk..kk+7, kk+16..kk+23}; lanes 16-31: +8 in K
        const v8h a0 = *(const v8h*)(arow + kk + hi * 8);
        const v8h a1 = *(const v8h*)(arow + kk + 16 + hi * 8);
        // B frag: column n / K {kk..kk+7, kk+16..kk+23}; lanes 16-31: +8 in K
        const v8h b0 = *(const v8h*)(brow + kk + hi * 8);
        const v8h b1 = *(const v8h*)(brow + kk + 16 + hi * 8);
        const v16h af = __builtin_shufflevector(a0, a1, 0,1,2,3,4,5,6,7,8,9,10,11,12,13,14,15);
        const v16h bf = __builtin_shufflevector(b0, b1, 0,1,2,3,4,5,6,7,8,9,10,11,12,13,14,15);
        acc = __builtin_amdgcn_wmma_f32_16x16x32_f16(false, af, false, bf,
                                                     (short)0, acc, false, false);
    }
    const int n = n0 + lo;
    if (n < N) {
        float bsv = 0.f;
        if (HAS_BIAS) bsv = bias[n];
#pragma unroll
        for (int r = 0; r < 8; r++) {
            const int m = m0 + hi * 8 + r;
            float v = acc[r] + bsv;
            if (ACT == 2) v = act_gelu(v);
            else if (ACT == 3) v = act_softplus(v);
            const long off = (long)m * ldc + n;
            if (HAS_RESID) v += resid[off];
            if (HAS_C)   C[off]   = v;
            if (HAS_C16) C16[off] = (_Float16)v;
        }
    }
}

// ---------------- f32 -> f16 convert, row-major, zero pad ----------------
__global__ void cvt_pad_kernel(const float* __restrict__ src, _Float16* __restrict__ dst,
                               int rows, int cols, int scols, int dcols, long total)
{
    long i = (long)blockIdx.x * blockDim.x + threadIdx.x;
    if (i >= total) return;
    int c = (int)(i % dcols);
    long r = i / dcols;
    float v = (r < rows && c < cols) ? src[r * (long)scols + c] : 0.f;
    dst[i] = (_Float16)v;
}

// ---------------- f32 -> f16 convert with transpose: dst[c*rowsPad + r] = src[r][c] ----------------
__global__ void cvt_pad_t_kernel(const float* __restrict__ src, _Float16* __restrict__ dst,
                                 int rows, int cols, int scols, int rowsPad, long total)
{
    long i = (long)blockIdx.x * blockDim.x + threadIdx.x;
    if (i >= total) return;
    int r = (int)(i % rowsPad);
    int c = (int)(i / rowsPad);
    float v = (r < rows && c < cols) ? src[(long)r * scols + c] : 0.f;
    dst[i] = (_Float16)v;
}

// ---------------- init: chan_in (B,T,128,32) -> x (B,T,32,128) ----------------
__global__ void init_x_kernel(const float* __restrict__ chan, float* __restrict__ x)
{
    long i = (long)blockIdx.x * blockDim.x + threadIdx.x;
    if (i >= (long)MTOK * D_MODEL) return;
    int d = (int)(i & 127);
    long r = i >> 7;
    int w = (int)(r & 31);
    long bt = r >> 5;
    x[i] = chan[(bt * D_MODEL + d) * WDIM + w];
}

// ---------------- attention: per (b,w,h,t) block, softmax over 127 keys ----------------
__global__ __launch_bounds__(128)
void attention_kernel(const float* __restrict__ q, const float* __restrict__ k,
                      const float* __restrict__ v, const unsigned char* __restrict__ mask,
                      _Float16* __restrict__ o16)
{
    const int t = blockIdx.x;       // 0..126
    const int w = blockIdx.y;       // 0..31
    const int bh = blockIdx.z;      // b*8+h
    const int b = bh >> 3, h = bh & 7;
    const int j = threadIdx.x;      // 0..127

    __shared__ float qs[16];
    __shared__ float red[128];
    __shared__ float sa[128];

    const long tokT = (((long)b * TLEN + t) * WDIM + w) * D_MODEL + h * 16;
    if (j < 16) qs[j] = q[tokT + j];
    __syncthreads();

    float s = -1e30f;
    if (j < TLEN) {
        const long tokJ = (((long)b * TLEN + j) * WDIM + w) * D_MODEL + h * 16;
        float acc = 0.f;
#pragma unroll
        for (int d = 0; d < 16; d++) acc += qs[d] * k[tokJ + d];
        s = acc * 0.25f;                       // 1/sqrt(16)
        if (mask[t * TLEN + j]) s = -1e30f;
    }
    red[j] = s; __syncthreads();
    for (int st = 64; st > 0; st >>= 1) { if (j < st) red[j] = fmaxf(red[j], red[j + st]); __syncthreads(); }
    const float mx = red[0]; __syncthreads();

    float e = (j < TLEN) ? __expf(s - mx) : 0.f;
    red[j] = e; __syncthreads();
    for (int st = 64; st > 0; st >>= 1) { if (j < st) red[j] += red[j + st]; __syncthreads(); }
    const float inv = 1.f / red[0]; __syncthreads();

    sa[j] = e * inv; __syncthreads();

    if (j < 16) {
        float acc = 0.f;
        for (int jj = 0; jj < TLEN; jj++) {
            const long tj = (((long)b * TLEN + jj) * WDIM + w) * D_MODEL + h * 16;
            acc += sa[jj] * v[tj + j];
        }
        o16[tokT + j] = (_Float16)acc;
    }
}

// ---------------- depthwise causal conv (k=4) + bias + SiLU ----------------
__global__ void conv_silu_kernel(const float* __restrict__ xz, const float* __restrict__ cw,
                                 const float* __restrict__ cb,
                                 float* __restrict__ xc32, _Float16* __restrict__ xc16)
{
    long i = (long)blockIdx.x * blockDim.x + threadIdx.x;
    if (i >= (long)MTOK * DI) return;
    int ch = (int)(i % DI);
    long row = i / DI;                 // b*LSEQ + l
    int l = (int)(row % LSEQ);
    long base = row - l;               // b*LSEQ
    float acc = cb[ch];
#pragma unroll
    for (int t2 = 0; t2 < 4; t2++) {
        int ll = l - 3 + t2;
        if (ll >= 0) acc += cw[t2 * DI + ch] * xz[(base + ll) * 512 + ch];
    }
    acc = act_silu(acc);
    xc32[i] = acc;
    xc16[i] = (_Float16)acc;
}

// ---------------- selective scan: one wave32 per (b,d); 2 states per lane ----------------
__global__ __launch_bounds__(32)
void mamba_scan_kernel(const float* __restrict__ delta, const float* __restrict__ xc32,
                       const float* __restrict__ dbl, const float* __restrict__ A_log,
                       float* __restrict__ ys)
{
    const int d = blockIdx.x;          // 0..255
    const int b = blockIdx.y;          // 0..3
    const int lane = threadIdx.x;      // 0..31
    const float A0 = -__expf(A_log[d * DSTATE + lane]);
    const float A1 = -__expf(A_log[d * DSTATE + lane + 32]);
    float h0 = 0.f, h1 = 0.f;
    for (int l = 0; l < LSEQ; l++) {
        const long row = (long)b * LSEQ + l;
        const float dt = delta[row * DI + d];
        const float xt = xc32[row * DI + d];
        const float* dr = dbl + row * 144;
        const float B0 = dr[DTRANK + lane],          B1 = dr[DTRANK + lane + 32];
        const float C0 = dr[DTRANK + DSTATE + lane], C1 = dr[DTRANK + DSTATE + lane + 32];
        const float u = dt * xt;
        h0 = h0 * __expf(dt * A0) + u * B0;
        h1 = h1 * __expf(dt * A1) + u * B1;
        float y = h0 * C0 + h1 * C1;
#pragma unroll
        for (int off = 16; off > 0; off >>= 1) y += __shfl_down(y, off, 32);
        if (lane == 0) ys[row * DI + d] = y;
    }
}

// ---------------- gate: (ys + xc*Dskip) * silu(z) -> f16 ----------------
__global__ void gate_kernel(const float* __restrict__ ys, const float* __restrict__ xc32,
                            const float* __restrict__ xz, const float* __restrict__ Dskip,
                            _Float16* __restrict__ yg16)
{
    long i = (long)blockIdx.x * blockDim.x + threadIdx.x;
    if (i >= (long)MTOK * DI) return;
    int ch = (int)(i % DI);
    long row = i / DI;
    float y = ys[i] + xc32[i] * Dskip[ch];
    float z = xz[row * 512 + DI + ch];
    yg16[i] = (_Float16)(y * act_silu(z));
}

// ---------------- fused residual-sum + LN(1e-5) -> x ; LN(1e-6) -> h16 ----------------
__global__ __launch_bounds__(128)
void ln_fused_kernel(float* __restrict__ x, const float* __restrict__ xa, const float* __restrict__ xm,
                     const float* __restrict__ g1, const float* __restrict__ be1,
                     const float* __restrict__ g2, const float* __restrict__ be2,
                     _Float16* __restrict__ h16)
{
    const int m = blockIdx.x;
    const int d = threadIdx.x;
    const long i = (long)m * D_MODEL + d;
    __shared__ float red[128];

    float v = x[i] + xa[i] + xm[i];
    red[d] = v; __syncthreads();
    for (int st = 64; st > 0; st >>= 1) { if (d < st) red[d] += red[d + st]; __syncthreads(); }
    const float mu = red[0] * (1.f / 128.f); __syncthreads();
    const float c = v - mu;
    red[d] = c * c; __syncthreads();
    for (int st = 64; st > 0; st >>= 1) { if (d < st) red[d] += red[d + st]; __syncthreads(); }
    const float var = red[0] * (1.f / 128.f); __syncthreads();
    const float xn = c * rsqrtf(var + 1e-5f) * g1[d] + be1[d];
    x[i] = xn;

    red[d] = xn; __syncthreads();
    for (int st = 64; st > 0; st >>= 1) { if (d < st) red[d] += red[d + st]; __syncthreads(); }
    const float mu2 = red[0] * (1.f / 128.f); __syncthreads();
    const float c2 = xn - mu2;
    red[d] = c2 * c2; __syncthreads();
    for (int st = 64; st > 0; st >>= 1) { if (d < st) red[d] += red[d + st]; __syncthreads(); }
    const float var2 = red[0] * (1.f / 128.f);
    h16[i] = (_Float16)(c2 * rsqrtf(var2 + 1e-6f) * g2[d] + be2[d]);
}

// ---------------- final gather: flat reinterpretation + index select ----------------
__global__ void gather_out_kernel(const float* __restrict__ x, const int* __restrict__ idx,
                                  float* __restrict__ out)
{
    const long TOT = 512L * 64 * 7 * 32;
    long o = (long)blockIdx.x * blockDim.x + threadIdx.x;
    if (o >= TOT) return;
    int j = (int)(o & 31);
    long r1 = o >> 5;
    int vv = (int)(r1 % 7);
    long r2 = r1 / 7;
    int n = (int)(r2 & 63);
    int r = (int)(r2 >> 6);                 // 0..511 (flat b*128+c)
    int l = idx[n * 7 + vv];
    long F = ((long)r * TLEN + l) * 32 + j; // flat pos in (B,W,T,D) order
    int d = (int)(F & 127); F >>= 7;
    int t = (int)(F % TLEN); F /= TLEN;
    int w = (int)(F & 31);
    int b = (int)(F >> 5);
    out[o] = x[(((long)b * TLEN + t) * WDIM + w) * D_MODEL + d];
}

// =====================================================================
extern "C" void kernel_launch(void* const* d_in, const int* in_sizes, int n_in,
                              void* d_out, int out_size, void* d_ws, size_t ws_size,
                              hipStream_t stream)
{
    (void)in_sizes; (void)n_in; (void)out_size; (void)ws_size;
    const float*         chan = (const float*)d_in[0];
    const unsigned char* mask = (const unsigned char*)d_in[1];
    const int*           idx  = (const int*)d_in[2];
    // params pytree leaves, sorted keys per layer:
    // 0 A_log 1 Dskip 2 Wk 3 Wo 4 Wq 5 Wv 6 b1 7 b2 8 bk 9 bo 10 bq 11 bv
    // 12 conv_b 13 conv_w 14 dt_b 15 dt_w 16 fln_b 17 fln_g 18 in_w 19 ln_b
    // 20 ln_g 21 out_w 22 w1 23 w2 24 xproj_w
    auto P = [&](int L, int k) -> const float* { return (const float*)d_in[3 + L * 25 + k]; };

    char* base = (char*)d_ws;
    size_t off = 0;
    auto allocF = [&](size_t n) -> float* {
        off = (off + 255) & ~(size_t)255; float* p = (float*)(base + off); off += n * sizeof(float); return p; };
    auto allocH = [&](size_t n) -> _Float16* {
        off = (off + 255) & ~(size_t)255; _Float16* p = (_Float16*)(base + off); off += n * sizeof(_Float16); return p; };

    const long M = MTOK;
    float*    x     = allocF(M * 128);
    float*    q     = allocF(M * 128);
    float*    k     = allocF(M * 128);
    float*    v     = allocF(M * 128);
    float*    xa    = allocF(M * 128);
    float*    xm    = allocF(M * 128);
    float*    xz    = allocF(M * 512);
    float*    xc32  = allocF(M * 256);
    float*    dbl   = allocF(M * 144);
    float*    delta = allocF(M * 256);
    float*    ys    = allocF(M * 256);
    _Float16* tok16 = allocH(M * 128);
    _Float16* o16   = allocH(M * 128);
    _Float16* xc16  = allocH(M * 256);
    _Float16* dt16  = allocH(M * 32);
    _Float16* yg16  = allocH(M * 256);
    _Float16* h16   = allocH(M * 128);
    _Float16* h1_16 = allocH(M * 128);
    // transposed f16 weights: Bt[n][k], size Np*Kp
    _Float16* wq16  = allocH(128 * 128);
    _Float16* wk16  = allocH(128 * 128);
    _Float16* wv16  = allocH(128 * 128);
    _Float16* wo16  = allocH(128 * 128);
    _Float16* inw16 = allocH(512 * 128);
    _Float16* xpw16 = allocH(144 * 256);
    _Float16* dtw16 = allocH(256 * 32);
    _Float16* ow16  = allocH(128 * 256);
    _Float16* w116  = allocH(128 * 128);
    _Float16* w216  = allocH(128 * 128);

    const int EB = 256;
    auto egrid = [&](long total) { return dim3((unsigned)((total + EB - 1) / EB)); };
    auto cvt = [&](const float* src, _Float16* dst, int rows, int cols, int scols, int dcols, int rowsPad) {
        long total = (long)rowsPad * dcols;
        cvt_pad_kernel<<<egrid(total), EB, 0, stream>>>(src, dst, rows, cols, scols, dcols, total);
    };
    // transpose convert: src[K x N] f32 -> dst[Npad x Kpad] f16 (N-major)
    auto cvtT = [&](const float* src, _Float16* dst, int K, int N, int Kpad, int Npad) {
        long total = (long)Npad * Kpad;
        cvt_pad_t_kernel<<<egrid(total), EB, 0, stream>>>(src, dst, K, N, N, Kpad, total);
    };

    // x = transpose(chan_in, (0,1,3,2))
    init_x_kernel<<<egrid(M * 128), EB, 0, stream>>>(chan, x);

    const dim3 gblk(128);
    auto ggrid = [&](int N) { return dim3((unsigned)((N + 15) / 16), (unsigned)((M + 63) / 64)); };

    for (int L = 0; L < 2; L++) {
        // ---- f16 conversions ----
        cvt(x, tok16, (int)M, 128, 128, 128, (int)M);
        cvtT(P(L, 4),  wq16,  128, 128, 128, 128);   // Wq
        cvtT(P(L, 2),  wk16,  128, 128, 128, 128);   // Wk
        cvtT(P(L, 5),  wv16,  128, 128, 128, 128);   // Wv
        cvtT(P(L, 3),  wo16,  128, 128, 128, 128);   // Wo
        cvtT(P(L, 18), inw16, 128, 512, 128, 512);   // in_w
        cvtT(P(L, 24), xpw16, 256, 136, 256, 144);   // xproj_w (pad N 136->144)
        cvtT(P(L, 15), dtw16,   8, 256,  32, 256);   // dt_w (pad K 8->32)
        cvtT(P(L, 21), ow16,  256, 128, 256, 128);   // out_w
        cvtT(P(L, 22), w116,  128, 128, 128, 128);   // w1
        cvtT(P(L, 23), w216,  128, 128, 128, 128);   // w2

        // ---- attention branch ----
        gemm_wmma_f16<0, true, false, true, false><<<ggrid(128), gblk, 0, stream>>>(
            tok16, wq16, P(L, 10), nullptr, q, nullptr, (int)M, 128, 128, 128, 128, 128);
        gemm_wmma_f16<0, true, false, true, false><<<ggrid(128), gblk, 0, stream>>>(
            tok16, wk16, P(L, 8),  nullptr, k, nullptr, (int)M, 128, 128, 128, 128, 128);
        gemm_wmma_f16<0, true, false, true, false><<<ggrid(128), gblk, 0, stream>>>(
            tok16, wv16, P(L, 11), nullptr, v, nullptr, (int)M, 128, 128, 128, 128, 128);
        attention_kernel<<<dim3(TLEN, WDIM, BATCH * 8), dim3(128), 0, stream>>>(q, k, v, mask, o16);
        gemm_wmma_f16<0, true, false, true, false><<<ggrid(128), gblk, 0, stream>>>(
            o16, wo16, P(L, 9), nullptr, xa, nullptr, (int)M, 128, 128, 128, 128, 128);

        // ---- mamba branch ----
        gemm_wmma_f16<0, false, false, true, false><<<ggrid(512), gblk, 0, stream>>>(
            tok16, inw16, nullptr, nullptr, xz, nullptr, (int)M, 512, 128, 128, 128, 512);
        conv_silu_kernel<<<egrid(M * 256), EB, 0, stream>>>(xz, P(L, 13), P(L, 12), xc32, xc16);
        gemm_wmma_f16<0, false, false, true, false><<<ggrid(144), gblk, 0, stream>>>(
            xc16, xpw16, nullptr, nullptr, dbl, nullptr, (int)M, 136, 256, 256, 256, 144);
        cvt(dbl, dt16, (int)M, 8, 144, 32, (int)M);  // pack dt cols, pad K->32
        gemm_wmma_f16<3, true, false, true, false><<<ggrid(256), gblk, 0, stream>>>(
            dt16, dtw16, P(L, 14), nullptr, delta, nullptr, (int)M, 256, 32, 32, 32, 256);
        mamba_scan_kernel<<<dim3(DI, BATCH), dim3(32), 0, stream>>>(delta, xc32, dbl, P(L, 0), ys);
        gate_kernel<<<egrid(M * 256), EB, 0, stream>>>(ys, xc32, xz, P(L, 1), yg16);
        gemm_wmma_f16<0, false, false, true, false><<<ggrid(128), gblk, 0, stream>>>(
            yg16, ow16, nullptr, nullptr, xm, nullptr, (int)M, 128, 256, 256, 256, 128);

        // ---- residual sum + LN(1e-5) -> x ; LN(1e-6) -> h16 ----
        ln_fused_kernel<<<dim3((unsigned)M), dim3(128), 0, stream>>>(
            x, xa, xm, P(L, 20), P(L, 19), P(L, 17), P(L, 16), h16);

        // ---- MLP with fused GELU and residual ----
        gemm_wmma_f16<2, true, false, false, true><<<ggrid(128), gblk, 0, stream>>>(
            h16, w116, P(L, 6), nullptr, nullptr, h1_16, (int)M, 128, 128, 128, 128, 128);
        gemm_wmma_f16<0, true, true, true, false><<<ggrid(128), gblk, 0, stream>>>(
            h1_16, w216, P(L, 7), x, x, nullptr, (int)M, 128, 128, 128, 128, 128);
    }

    // ---- output gather ----
    gather_out_kernel<<<egrid(512L * 64 * 7 * 32), EB, 0, stream>>>(x, idx, (float*)d_out);
}
